// ModelStage_77833397338614
// MI455X (gfx1250) — compile-verified, matrix operand order checked by reading, not run on hardware
//
#include <hip/hip_runtime.h>
#include <hip/hip_bf16.h>

typedef __attribute__((ext_vector_type(16))) _Float16 v16h;
typedef __attribute__((ext_vector_type(8)))  float    v8f;

#define BB   4
#define CH   64
#define HH   128
#define WW   128
#define HW   (HH*WW)
#define NPIX (BB*HW)        // 65536 pixels
#define PNUM 63
#define KS   5
#define KK   25             // 5*5 taps, padded to K=32 for WMMA

// Fragment slot -> K index for the 16-bit 16x32 A/B WMMA layout.
__device__ __forceinline__ int frag_k(int r, int j, int hi) {
    return 2*(r & 3) + hi*8 + (r >> 2)*16 + j;
}

// ---------------------------------------------------------------------------
// Kernel A: conv0 (1->64, 5x5, replicate pad) via WMMA implicit GEMM,
// fused RBF activation producing a0 = act(c0,w0) and g0 = act'(c0,w0).
// One wave -> 16 pixels x 64 channels (4 WMMAs sharing one A fragment).
// Weights staged in *fragment order* so each lane loads its v16h with two
// contiguous ds_load_b128.
// ---------------------------------------------------------------------------
__global__ void __launch_bounds__(128) conv0_act_kernel(
    const float* __restrict__ x,  const float* __restrict__ f0,
    const float* __restrict__ b0, const float* __restrict__ w0,
    float* __restrict__ a0, float* __restrict__ g0)
{
    __shared__ __align__(32) _Float16 patch[64][32];      // pixels x K
    __shared__ __align__(32) _Float16 wfrag[2][64][16];   // [hi][n][slot]
    __shared__ float w0s[CH*PNUM];
    __shared__ float bs[CH];

    const int tid = threadIdx.x;
    const int pixelBase = blockIdx.x * 64;

    // Stage weights directly in per-lane fragment order
    for (int i = tid; i < 2*64*16; i += 128) {
        int slot = i & 15, n = (i >> 4) & 63, hig = i >> 10;
        int k = frag_k(slot >> 1, slot & 1, hig);
        float v = (k < KK) ? f0[n*KK + k] : 0.0f;
        wfrag[hig][n][slot] = (_Float16)v;
    }
    for (int i = tid; i < CH*PNUM; i += 128) w0s[i] = w0[i];
    if (tid < CH) bs[tid] = b0[tid];

    // Stage input patches (replicate padding), scaled by 255
    for (int i = tid; i < 64*32; i += 128) {
        int p = i >> 5, k = i & 31;
        int pixel = pixelBase + p;
        int b = pixel / HW, rem = pixel % HW;
        int h = rem / WW,  w = rem % WW;
        float v = 0.0f;
        if (k < KK) {
            int kh = k / KS, kw = k % KS;
            int ih = min(max(h + kh - 2, 0), HH-1);
            int iw = min(max(w + kw - 2, 0), WW-1);
            v = x[b*HW + ih*WW + iw] * 255.0f;
        }
        patch[p][k] = (_Float16)v;
    }
    // Prefetch next tile's input while this block computes
    if (tid == 0 && (size_t)(pixelBase + 64) < (size_t)NPIX)
        __builtin_prefetch((const void*)(x + pixelBase + 64), 0, 1);
    __syncthreads();

    const int wave = tid >> 5;
    const int lane = tid & 31;
    const int m16  = lane & 15;
    const int hi   = lane >> 4;      // lane group 0/1

    // A fragment: two contiguous 8-half runs -> 2x ds_load_b128
    v16h afrag;
    const int arow = wave*16 + m16;
    #pragma unroll
    for (int r = 0; r < 8; ++r) {
        int k0 = frag_k(r, 0, hi);
        afrag[2*r]   = patch[arow][k0];
        afrag[2*r+1] = patch[arow][k0+1];
    }

    #pragma unroll
    for (int t = 0; t < 4; ++t) {
        const int n = t*16 + m16;
        v16h bfrag = *(const v16h*)&wfrag[hi][n][0];   // one 32B LDS fetch
        v8f c = {};
        c = __builtin_amdgcn_wmma_f32_16x16x32_f16(false, afrag, false, bfrag,
                                                   (short)0, c, false, false);
        // Epilogue: bias + fused RBF value/gradient (one exp stream for both)
        const float bias = bs[n];
        const float* wp = &w0s[n*PNUM];
        #pragma unroll
        for (int v = 0; v < 8; ++v) {
            int M = v + hi*8;                     // 32-bit C/D layout
            int pixel = pixelBase + wave*16 + M;
            int b = pixel / HW, rem = pixel % HW;
            float s  = c[v] + bias;
            float d  = s + 310.0f;                // s - mu_0, mu_p step = 10
            float av = 0.0f, gv = 0.0f;
            for (int p = 0; p < PNUM; ++p) {
                float e  = __expf(d*d * (-0.005f));   // exp(-d^2/200)
                float wv = wp[p];
                av += wv * e;
                gv += wv * e * d;
                d  -= 10.0f;
            }
            gv *= -0.01f;                              // * (d/-100)
            size_t o = ((size_t)b*CH + n)*HW + rem;
            a0[o] = av;
            g0[o] = gv;
        }
    }
}

// ---------------------------------------------------------------------------
// Kernel B: conv1 (64->1, 5x5, replicate pad) + RBF(w1) -> a1.  N=1 so WMMA
// is useless; VALU dot products out of L2 (a0 fits easily in 192MB L2).
// ---------------------------------------------------------------------------
__global__ void __launch_bounds__(256) conv1_act_kernel(
    const float* __restrict__ a0, const float* __restrict__ f1,
    const float* __restrict__ b1, const float* __restrict__ w1,
    float* __restrict__ a1)
{
    __shared__ float f1s[CH*KK];
    __shared__ float w1s[PNUM];
    const int tid = threadIdx.x;
    for (int i = tid; i < CH*KK; i += 256) f1s[i] = f1[i];
    if (tid < PNUM) w1s[tid] = w1[tid];
    __syncthreads();

    const int pixel = blockIdx.x*256 + tid;
    const int b = pixel / HW, rem = pixel % HW;
    const int h = rem / WW,  w = rem % WW;

    float acc = b1[0];
    for (int c = 0; c < CH; ++c) {
        const float* ap = a0 + ((size_t)b*CH + c)*HW;
        #pragma unroll
        for (int kh = 0; kh < KS; ++kh) {
            int ih = min(max(h + kh - 2, 0), HH-1);
            #pragma unroll
            for (int kw = 0; kw < KS; ++kw) {
                int iw = min(max(w + kw - 2, 0), WW-1);
                acc += ap[ih*WW + iw] * f1s[c*KK + kh*KS + kw];
            }
        }
    }
    float av = 0.0f;
    float d  = acc + 310.0f;
    for (int p = 0; p < PNUM; ++p) {
        av += w1s[p] * __expf(d*d * (-0.005f));
        d  -= 10.0f;
    }
    a1[pixel] = av;
}

// ---------------------------------------------------------------------------
// Kernel C: crop(conv_transpose(a1, f1)) = zero-padded conv with flipped f1,
// 1->64 channels, via WMMA; fused elementwise multiply by g0 -> d.
// ---------------------------------------------------------------------------
__global__ void __launch_bounds__(128) deconv1_mul_kernel(
    const float* __restrict__ a1, const float* __restrict__ f1,
    const float* __restrict__ g0, float* __restrict__ dbuf)
{
    __shared__ __align__(32) _Float16 patch[64][32];
    __shared__ __align__(32) _Float16 wfrag[2][64][16];

    const int tid = threadIdx.x;
    const int pixelBase = blockIdx.x * 64;

    // Flipped weights in fragment order: wfrag[hi][n][slot] = f1[n][4-kh][4-kw]
    for (int i = tid; i < 2*64*16; i += 128) {
        int slot = i & 15, n = (i >> 4) & 63, hig = i >> 10;
        int k = frag_k(slot >> 1, slot & 1, hig);
        float v = 0.0f;
        if (k < KK) {
            int kh = k / KS, kw = k % KS;
            v = f1[n*KK + (4-kh)*KS + (4-kw)];
        }
        wfrag[hig][n][slot] = (_Float16)v;
    }
    // Patches from a1 with ZERO padding (transpose-conv boundary)
    for (int i = tid; i < 64*32; i += 128) {
        int p = i >> 5, k = i & 31;
        int pixel = pixelBase + p;
        int b = pixel / HW, rem = pixel % HW;
        int h = rem / WW,  w = rem % WW;
        float v = 0.0f;
        if (k < KK) {
            int kh = k / KS, kw = k % KS;
            int ih = h + kh - 2, iw = w + kw - 2;
            if (ih >= 0 && ih < HH && iw >= 0 && iw < WW)
                v = a1[b*HW + ih*WW + iw];
        }
        patch[p][k] = (_Float16)v;
    }
    if (tid == 0 && (size_t)(pixelBase + 64) < (size_t)NPIX)
        __builtin_prefetch((const void*)(a1 + pixelBase + 64), 0, 1);
    __syncthreads();

    const int wave = tid >> 5;
    const int lane = tid & 31;
    const int m16  = lane & 15;
    const int hi   = lane >> 4;

    v16h afrag;
    const int arow = wave*16 + m16;
    #pragma unroll
    for (int r = 0; r < 8; ++r) {
        int k0 = frag_k(r, 0, hi);
        afrag[2*r]   = patch[arow][k0];
        afrag[2*r+1] = patch[arow][k0+1];
    }

    #pragma unroll
    for (int t = 0; t < 4; ++t) {
        const int n = t*16 + m16;
        v16h bfrag = *(const v16h*)&wfrag[hi][n][0];
        v8f c = {};
        c = __builtin_amdgcn_wmma_f32_16x16x32_f16(false, afrag, false, bfrag,
                                                   (short)0, c, false, false);
        #pragma unroll
        for (int v = 0; v < 8; ++v) {
            int M = v + hi*8;
            int pixel = pixelBase + wave*16 + M;
            int b = pixel / HW, rem = pixel % HW;
            size_t o = ((size_t)b*CH + n)*HW + rem;
            dbuf[o] = c[v] * g0[o];
        }
    }
}

// ---------------------------------------------------------------------------
// Kernel D: crop(conv_transpose(d, f0)) (64->1, zero pad, flipped f0) + final
// residual combine:  out = (x*255 - (dd + exp(lam)*(x-y)*255)) / 255
// ---------------------------------------------------------------------------
__global__ void __launch_bounds__(256) deconv0_out_kernel(
    const float* __restrict__ dbuf, const float* __restrict__ f0,
    const float* __restrict__ x, const float* __restrict__ y,
    const float* __restrict__ lam_param, float* __restrict__ out)
{
    __shared__ float f0s[CH*KK];   // pre-flipped
    const int tid = threadIdx.x;
    for (int i = tid; i < CH*KK; i += 256) {
        int c = i / KK, k = i % KK;
        int kh = k / KS, kw = k % KS;
        f0s[i] = f0[c*KK + (4-kh)*KS + (4-kw)];
    }
    __syncthreads();

    const int pixel = blockIdx.x*256 + tid;
    const int b = pixel / HW, rem = pixel % HW;
    const int h = rem / WW,  w = rem % WW;

    float acc = 0.0f;
    for (int c = 0; c < CH; ++c) {
        const float* dp = dbuf + ((size_t)b*CH + c)*HW;
        #pragma unroll
        for (int kh = 0; kh < KS; ++kh) {
            int ih = h + kh - 2;
            if (ih < 0 || ih >= HH) continue;
            #pragma unroll
            for (int kw = 0; kw < KS; ++kw) {
                int iw = w + kw - 2;
                if (iw < 0 || iw >= WW) continue;
                acc += dp[ih*WW + iw] * f0s[c*KK + kh*KS + kw];
            }
        }
    }
    const float el = __expf(lam_param[0]);
    const float xv = x[pixel]*255.0f, yv = y[pixel]*255.0f;
    out[pixel] = (xv - (acc + el*(xv - yv))) * (1.0f/255.0f);
}

// ---------------------------------------------------------------------------
extern "C" void kernel_launch(void* const* d_in, const int* in_sizes, int n_in,
                              void* d_out, int out_size, void* d_ws, size_t ws_size,
                              hipStream_t stream) {
    (void)in_sizes; (void)n_in; (void)out_size; (void)ws_size;
    // Inputs (setup_inputs order): x, y, lam(int,ignored), f0, b0, f1, b1, w0, w1, lam_param
    const float* x   = (const float*)d_in[0];
    const float* y   = (const float*)d_in[1];
    const float* f0  = (const float*)d_in[3];
    const float* b0  = (const float*)d_in[4];
    const float* f1  = (const float*)d_in[5];
    const float* b1  = (const float*)d_in[6];
    const float* w0  = (const float*)d_in[7];
    const float* w1  = (const float*)d_in[8];
    const float* lam = (const float*)d_in[9];
    float* out = (float*)d_out;

    // Workspace layout: a0 (16MB) | g0 (16MB) | a1 (256KB); d reuses a0 region
    char*  ws = (char*)d_ws;
    float* a0 = (float*)ws;
    float* g0 = (float*)(ws + (size_t)NPIX*CH*sizeof(float));
    float* a1 = (float*)(ws + (size_t)2*NPIX*CH*sizeof(float));
    float* db = a0;   // a0 is dead after conv1_act; reuse for d

    conv0_act_kernel  <<<NPIX/64, 128, 0, stream>>>(x, f0, b0, w0, a0, g0);
    conv1_act_kernel  <<<NPIX/256, 256, 0, stream>>>(a0, f1, b1, w1, a1);
    deconv1_mul_kernel<<<NPIX/64, 128, 0, stream>>>(a1, f1, g0, db);
    deconv0_out_kernel<<<NPIX/256, 256, 0, stream>>>(db, f0, x, y, lam, out);
}
